// CrossTemporalAttention_16501264351852
// MI455X (gfx1250) — 16-bit and 8-bit matrix operand order repaired and checked by reading, compiled, not run on hardware
//
#include <hip/hip_runtime.h>

typedef __attribute__((ext_vector_type(16))) __bf16 v16bf;
typedef __attribute__((ext_vector_type(16))) int    v16i;
typedef __attribute__((ext_vector_type(8)))  float  v8f;
typedef __bf16 bf16_t;

union BF16x16 {
    v16bf v;
    uint4 u4[2];
};
union I16 {
    v16i  v;
    unsigned u[16];
    uint4 q[4];
};

#define BB   4
#define CC   256
#define C8K  32
#define NN   4096

#define LOG2E 1.4426950408889634f

__device__ __forceinline__ v8f wmma_bf16(const v16bf& a, const v16bf& b, const v8f& c) {
    return __builtin_amdgcn_wmma_f32_16x16x32_bf16(false, a, false, b, (short)0, c,
                                                   false, false);
}
__device__ __forceinline__ v8f wmma_fp8(const v16i& a, const v16i& b, const v8f& c) {
    return __builtin_amdgcn_wmma_f32_16x16x128_fp8_fp8(a, b, (short)0, c, false, false);
}

#if __has_builtin(__builtin_amdgcn_exp2f)
__device__ __forceinline__ float fexp2(float x) { return __builtin_amdgcn_exp2f(x); }
#else
__device__ __forceinline__ float fexp2(float x) { return exp2f(x); }
#endif

#if __has_builtin(__builtin_amdgcn_perm)
__device__ __forceinline__ unsigned pack_bf16(float a, float b) {
    return __builtin_amdgcn_perm(__float_as_uint(b), __float_as_uint(a), 0x07060302u);
}
#else
__device__ __forceinline__ unsigned pack_bf16(float a, float b) {
    return (__float_as_uint(a) >> 16) | (__float_as_uint(b) & 0xFFFF0000u);
}
#endif
__device__ __forceinline__ float unpack_lo(unsigned u) { return __uint_as_float(u << 16); }
__device__ __forceinline__ float unpack_hi(unsigned u) { return __uint_as_float(u & 0xFFFF0000u); }

__device__ __forceinline__ unsigned f32_to_fp8_bits(float x) {
    unsigned u  = __float_as_uint(x);
    unsigned s  = (u >> 24) & 0x80u;
    unsigned au = u & 0x7FFFFFFFu;
    if (au > 0x43E00000u) au = 0x43E00000u;
    int e = (int)(au >> 23) - 120;
    unsigned m = (au >> 20) & 7u;
    unsigned body = (e <= 0) ? 0u : ((((unsigned)e << 3) | m) + ((au >> 19) & 1u));
    if (body > 0x7Eu) body = 0x7Eu;
    return s | body;
}
#if __has_builtin(__builtin_amdgcn_cvt_pk_fp8_f32)
__device__ __forceinline__ unsigned pack_fp8x4(float a, float b, float c, float d) {
    int v = __builtin_amdgcn_cvt_pk_fp8_f32(a, b, 0, false);
    v     = __builtin_amdgcn_cvt_pk_fp8_f32(c, d, v, true);
    return (unsigned)v;
}
__device__ __forceinline__ unsigned char f32_to_fp8(float x) {
    return (unsigned char)(__builtin_amdgcn_cvt_pk_fp8_f32(x, x, 0, false) & 0xFF);
}
#else
__device__ __forceinline__ unsigned pack_fp8x4(float a, float b, float c, float d) {
    return f32_to_fp8_bits(a) | (f32_to_fp8_bits(b) << 8) |
           (f32_to_fp8_bits(c) << 16) | (f32_to_fp8_bits(d) << 24);
}
__device__ __forceinline__ unsigned char f32_to_fp8(float x) {
    return (unsigned char)f32_to_fp8_bits(x);
}
#endif

__global__ __launch_bounds__(256) void cta_cvt_kernel(
    const float* __restrict__ pre, const float* __restrict__ post,
    const float* __restrict__ Wq, const float* __restrict__ Wk,
    const float* __restrict__ Wv,
    bf16_t* __restrict__ preTB, bf16_t* __restrict__ postTB,
    bf16_t* __restrict__ WqB, bf16_t* __restrict__ WkB, bf16_t* __restrict__ WvB)
{
    __shared__ float sT[32][33];
    const int TT  = 2 * BB * (CC / 32) * (NN / 32);
    const int blk = blockIdx.x;
    const int t   = threadIdx.x;

    if (blk < TT) {
        const int which = blk & 1;
        const int tb    = blk >> 1;
        const int tilesPerB = (CC / 32) * (NN / 32);
        const int b   = tb / tilesPerB;
        const int rem = tb % tilesPerB;
        const int c0  = (rem / (NN / 32)) * 32;
        const int n0  = (rem % (NN / 32)) * 32;
        const float* src = which ? post : pre;
        bf16_t*      dst = which ? postTB : preTB;
        const int tl = t & 31, th = t >> 5;
#pragma unroll
        for (int rr = 0; rr < 4; ++rr) {
            const int cl = th + rr * 8;
            sT[cl][tl] = src[((size_t)(b * CC) + c0 + cl) * NN + n0 + tl];
        }
        __syncthreads();
#pragma unroll
        for (int rr = 0; rr < 4; ++rr) {
            const int nl = th + rr * 8;
            dst[((size_t)(b * NN) + n0 + nl) * CC + c0 + tl] = (bf16_t)sT[tl][nl];
        }
    } else {
        const int idx = (blk - TT) * 1024 + t * 4;
#pragma unroll
        for (int u = 0; u < 4; ++u) {
            const int i = idx + u;
            if (i < 8192)       WqB[i]         = (bf16_t)Wq[i];
            else if (i < 16384) WkB[i - 8192]  = (bf16_t)Wk[i - 8192];
            else                WvB[i - 16384] = (bf16_t)Wv[i - 16384];
        }
    }
}

__global__ __launch_bounds__(256) void cta_projv_kernel(
    const bf16_t* __restrict__ preTB, const bf16_t* __restrict__ WvB,
    const float* __restrict__ bv, unsigned char* __restrict__ vm8)
{
    const int lane = threadIdx.x & 31;
    const int wid  = blockIdx.x * 8 + (threadIdx.x >> 5);
    const int perB = (CC / 16) * (NN / 64);
    const int b    = wid / perB;
    const int rem  = wid % perB;
    const int c0   = (rem / (NN / 64)) * 16;
    const int n0   = (rem % (NN / 64)) * 64;
    const int half = lane >> 4, l15 = lane & 15;
    const bf16_t* pT = preTB + (size_t)b * NN * CC;

    v8f acc[4];
#pragma unroll
    for (int u = 0; u < 4; ++u)
#pragma unroll
        for (int r = 0; r < 8; ++r) acc[u][r] = 0.0f;

    for (int kk = 0; kk < CC; kk += 32) {
        BF16x16 Wa;
        const bf16_t* wp = WvB + (size_t)(c0 + l15) * CC + kk + half * 16;
        Wa.u4[0] = *(const uint4*)(wp);
        Wa.u4[1] = *(const uint4*)(wp + 8);
#pragma unroll
        for (int u = 0; u < 4; ++u) {
            BF16x16 Bb;
            Bb.v = *(const v16bf*)(pT + (size_t)(n0 + u * 16 + l15) * CC + kk + half * 16);
            acc[u] = wmma_bf16(Wa.v, Bb.v, acc[u]);
        }
    }
#pragma unroll
    for (int r = 0; r < 8; ++r) {
        const int c = c0 + r + 8 * half;
        const float bb = bv[c];
#pragma unroll
        for (int u = 0; u < 4; ++u)
            vm8[((size_t)b * CC + c) * NN + n0 + u * 16 + l15] = f32_to_fp8(acc[u][r] + bb);
    }
}

__global__ __launch_bounds__(256) void cta_projqk_kernel(
    const bf16_t* __restrict__ preTB, const bf16_t* __restrict__ postTB,
    const bf16_t* __restrict__ WqB, const bf16_t* __restrict__ WkB,
    const float* __restrict__ bq, const float* __restrict__ bk,
    bf16_t* __restrict__ qT, bf16_t* __restrict__ kT)
{
    const int lane  = threadIdx.x & 31;
    const int wid   = blockIdx.x * 8 + (threadIdx.x >> 5);
    const int which = wid & 1;
    const int rem   = wid >> 1;
    const int b     = rem / (NN / 32);
    const int n0    = (rem % (NN / 32)) * 32;
    const int half  = lane >> 4, l15 = lane & 15;

    const bf16_t* src  = which ? preTB : postTB;
    const bf16_t* W    = which ? WkB : WqB;
    const float*  bias = which ? bk : bq;
    bf16_t*       dst  = which ? kT : qT;
    const bf16_t* sT   = src + (size_t)b * NN * CC;

    v8f acc[2][2];
#pragma unroll
    for (int ta = 0; ta < 2; ++ta)
#pragma unroll
        for (int tn = 0; tn < 2; ++tn)
#pragma unroll
            for (int r = 0; r < 8; ++r) acc[ta][tn][r] = 0.0f;

    for (int kk = 0; kk < CC; kk += 32) {
        BF16x16 A[2], Bt[2];
#pragma unroll
        for (int ta = 0; ta < 2; ++ta) {
            const bf16_t* wp = W + (size_t)(ta * 16 + l15) * CC + kk + half * 16;
            A[ta].u4[0] = *(const uint4*)(wp);
            A[ta].u4[1] = *(const uint4*)(wp + 8);
        }
#pragma unroll
        for (int tn = 0; tn < 2; ++tn)
            Bt[tn].v = *(const v16bf*)(sT + (size_t)(n0 + tn * 16 + l15) * CC + kk + half * 16);
#pragma unroll
        for (int ta = 0; ta < 2; ++ta)
#pragma unroll
            for (int tn = 0; tn < 2; ++tn)
                acc[ta][tn] = wmma_bf16(A[ta].v, Bt[tn].v, acc[ta][tn]);
    }
#pragma unroll
    for (int ta = 0; ta < 2; ++ta)
#pragma unroll
        for (int r = 0; r < 8; ++r) {
            const int d = ta * 16 + r + 8 * half;
            const float bb = bias[d];
#pragma unroll
            for (int tn = 0; tn < 2; ++tn)
                dst[((size_t)b * NN + n0 + tn * 16 + l15) * C8K + d] =
                    (bf16_t)(acc[ta][tn][r] + bb);
        }
}

__global__ __launch_bounds__(256) void cta_flash_kernel(
    const bf16_t* __restrict__ qT, const bf16_t* __restrict__ kT,
    const unsigned char* __restrict__ vm8, const float* __restrict__ post_feat,
    const float* __restrict__ gamma, float* __restrict__ out)
{
    const int lane = threadIdx.x & 31;
    const int wid  = blockIdx.x * 8 + (threadIdx.x >> 5);
    const int h    = wid & 1;
    const int g    = wid >> 1;
    const int b    = g >> 7;
    const int i0   = (g & 127) << 5;
    const int half = lane >> 4, l15 = lane & 15;

    const bf16_t* qTb = qT + (size_t)b * NN * C8K;
    const unsigned char* kbase =
        (const unsigned char*)(kT + (size_t)b * NN * C8K + (size_t)l15 * C8K + half * 16);
    const unsigned char* vbase =
        vm8 + ((size_t)b * CC + h * 128 + l15) * NN + half * 8;

    BF16x16 Qb0, Qb1;
    Qb0.v = *(const v16bf*)(qTb + (size_t)(i0 + l15) * C8K + half * 16);
    Qb1.v = *(const v16bf*)(qTb + (size_t)(i0 + 16 + l15) * C8K + half * 16);

    v8f acc0[8], acc1[8];
#pragma unroll
    for (int tt = 0; tt < 8; ++tt)
#pragma unroll
        for (int r = 0; r < 8; ++r) { acc0[tt][r] = 0.0f; acc1[tt][r] = 0.0f; }

    float M0 = -1e30f, L0 = 0.0f, M1 = -1e30f, L1 = 0.0f;
    const v8f zero8 = {0.f, 0.f, 0.f, 0.f, 0.f, 0.f, 0.f, 0.f};

    for (int j0 = 0; j0 < NN; j0 += 128) {
        const unsigned char* kp = kbase + (size_t)j0 * (C8K * 2);
        const unsigned char* vp = vbase + j0;

        unsigned Sb0[8][4], Sb1[8][4];
        float m0 = -1e30f, m1 = -1e30f;
#pragma unroll
        for (int jt = 0; jt < 8; ++jt) {
            BF16x16 Ka;
            Ka.u4[0] = *(const uint4*)(kp + jt * 1024);
            Ka.u4[1] = *(const uint4*)(kp + jt * 1024 + 16);
            const v8f St0 = wmma_bf16(Ka.v, Qb0.v, zero8);
            const v8f St1 = wmma_bf16(Ka.v, Qb1.v, zero8);
#pragma unroll
            for (int r = 0; r < 8; ++r) {
                m0 = fmaxf(m0, St0[r]);
                m1 = fmaxf(m1, St1[r]);
            }
#pragma unroll
            for (int p = 0; p < 4; ++p) {
                Sb0[jt][p] = pack_bf16(St0[2 * p], St0[2 * p + 1]);
                Sb1[jt][p] = pack_bf16(St1[2 * p], St1[2 * p + 1]);
            }
        }
        m0 = fmaxf(m0, __shfl_xor(m0, 16, 32));
        m1 = fmaxf(m1, __shfl_xor(m1, 16, 32));
        const float Mn0 = fmaxf(M0, m0), Mn1 = fmaxf(M1, m1);
        const float alpha0 = fexp2((M0 - Mn0) * LOG2E);
        const float alpha1 = fexp2((M1 - Mn1) * LOG2E);
        M0 = Mn0; M1 = Mn1;
        const float nM0 = -Mn0 * LOG2E, nM1 = -Mn1 * LOG2E;

        float s0 = 0.f, s1 = 0.f;
        I16 Pa0, Pa1;
#pragma unroll
        for (int jt = 0; jt < 8; ++jt) {
            float p0[8], p1[8];
#pragma unroll
            for (int p = 0; p < 4; ++p) {
                p0[2 * p]     = fexp2(fmaf(unpack_lo(Sb0[jt][p]), LOG2E, nM0));
                p0[2 * p + 1] = fexp2(fmaf(unpack_hi(Sb0[jt][p]), LOG2E, nM0));
                p1[2 * p]     = fexp2(fmaf(unpack_lo(Sb1[jt][p]), LOG2E, nM1));
                p1[2 * p + 1] = fexp2(fmaf(unpack_hi(Sb1[jt][p]), LOG2E, nM1));
            }
#pragma unroll
            for (int r = 0; r < 8; ++r) { s0 += p0[r]; s1 += p1[r]; }
            Pa0.u[2 * jt]     = pack_fp8x4(p0[0], p0[1], p0[2], p0[3]);
            Pa0.u[2 * jt + 1] = pack_fp8x4(p0[4], p0[5], p0[6], p0[7]);
            Pa1.u[2 * jt]     = pack_fp8x4(p1[0], p1[1], p1[2], p1[3]);
            Pa1.u[2 * jt + 1] = pack_fp8x4(p1[4], p1[5], p1[6], p1[7]);
        }
        s0 += __shfl_xor(s0, 16, 32);
        s1 += __shfl_xor(s1, 16, 32);
        L0 = L0 * alpha0 + s0;
        L1 = L1 * alpha1 + s1;

        float aR0[8], aR1[8];
#pragma unroll
        for (int r = 0; r < 8; ++r) {
            aR0[r] = __shfl(alpha0, r + 8 * half, 32);
            aR1[r] = __shfl(alpha1, r + 8 * half, 32);
        }

#pragma unroll
        for (int tt = 0; tt < 8; ++tt) {
            I16 Vb;
#pragma unroll
            for (int t = 0; t < 8; ++t) {
                const uint2 w = *(const uint2*)(vp + tt * 16 * NN + 16 * t);
                Vb.u[2 * t]     = w.x;
                Vb.u[2 * t + 1] = w.y;
            }
#pragma unroll
            for (int r = 0; r < 8; ++r) { acc0[tt][r] *= aR0[r]; acc1[tt][r] *= aR1[r]; }
            acc0[tt] = wmma_fp8(Pa0.v, Vb.v, acc0[tt]);
            acc1[tt] = wmma_fp8(Pa1.v, Vb.v, acc1[tt]);
        }
    }

    const float gmm = gamma[0];
    float inv0[8], inv1[8];
#pragma unroll
    for (int r = 0; r < 8; ++r) {
        inv0[r] = 1.0f / __shfl(L0, r + 8 * half, 32);
        inv1[r] = 1.0f / __shfl(L1, r + 8 * half, 32);
    }
#pragma unroll
    for (int tt = 0; tt < 8; ++tt) {
        const int c = h * 128 + tt * 16 + l15;
        const float* pf = post_feat + ((size_t)b * CC + c) * NN + i0;
        float*       op = out       + ((size_t)b * CC + c) * NN + i0;
#pragma unroll
        for (int r = 0; r < 8; ++r) {
            const int ia = r + 8 * half;
            op[ia]      = gmm * acc0[tt][r] * inv0[r] + pf[ia];
            op[16 + ia] = gmm * acc1[tt][r] * inv1[r] + pf[16 + ia];
        }
    }
}

extern "C" void kernel_launch(void* const* d_in, const int* in_sizes, int n_in,
                              void* d_out, int out_size, void* d_ws, size_t ws_size,
                              hipStream_t stream) {
    const float* pre   = (const float*)d_in[0];
    const float* post  = (const float*)d_in[1];
    const float* Wq    = (const float*)d_in[2];
    const float* bq    = (const float*)d_in[3];
    const float* Wk    = (const float*)d_in[4];
    const float* bk    = (const float*)d_in[5];
    const float* Wv    = (const float*)d_in[6];
    const float* bv    = (const float*)d_in[7];
    const float* gamma = (const float*)d_in[8];
    float* out = (float*)d_out;

    bf16_t* preTB  = (bf16_t*)d_ws;
    bf16_t* postTB = preTB  + (size_t)BB * NN * CC;
    bf16_t* qT     = postTB + (size_t)BB * NN * CC;
    bf16_t* kT     = qT     + (size_t)BB * NN * C8K;
    unsigned char* vm8 = (unsigned char*)(kT + (size_t)BB * NN * C8K);
    bf16_t* WqB    = (bf16_t*)(vm8 + (size_t)BB * CC * NN);
    bf16_t* WkB    = WqB + (size_t)C8K * CC;
    bf16_t* WvB    = WkB + (size_t)C8K * CC;

    const int cvtBlocks = 2 * BB * (CC / 32) * (NN / 32) + 80;
    cta_cvt_kernel<<<cvtBlocks, 256, 0, stream>>>(
        pre, post, Wq, Wk, Wv, preTB, postTB, WqB, WkB, WvB);

    cta_projv_kernel<<<(BB * (CC / 16) * (NN / 64)) / 8, 256, 0, stream>>>(
        preTB, WvB, bv, vm8);

    cta_projqk_kernel<<<(BB * (NN / 32) * 2) / 8, 256, 0, stream>>>(
        preTB, postTB, WqB, WkB, bq, bk, qT, kT);

    cta_flash_kernel<<<(BB * (NN / 32) * 2) / 8, 256, 0, stream>>>(
        qT, kT, vm8, post, gamma, out);
}
